// Student_43293270344104
// MI455X (gfx1250) — matrix operand order repaired and checked by reading, not compiled, not run on hardware
//
#include <hip/hip_runtime.h>
#include <hip/hip_bf16.h>
#include <math.h>

// ---------------- problem constants (from reference) ----------------
#define B_   64
#define C_   3
#define IMG_ 224
#define P_   16
#define D_   768
#define NH_  12
#define NL_  12
#define G_   (IMG_ / P_)
#define T_   (G_ * G_)
#define HD_  64            // D / NH
#define EPS_ 1e-5f
#define SMAX 224           // max padded context length (T=196 -> ceil32 = 224)

// ---------------- types ----------------
typedef __bf16 bf16;
typedef __attribute__((ext_vector_type(16))) __bf16 v16bf;
typedef __attribute__((ext_vector_type(8)))  __bf16 v8bf;
typedef __attribute__((ext_vector_type(8)))  float  v8f;

static __device__ __forceinline__ v16bf cat8(v8bf lo, v8bf hi) {
  return __builtin_shufflevector(lo, hi, 0,1,2,3,4,5,6,7,8,9,10,11,12,13,14,15);
}
static __device__ __forceinline__ unsigned short f2bf(float f) {
  union { float f; unsigned int u; } a; a.f = f;
  unsigned int r = a.u + 0x7FFFu + ((a.u >> 16) & 1u);   // round-to-nearest-even
  return (unsigned short)(r >> 16);
}
static __device__ __forceinline__ bf16 bfbits(unsigned short s) {
  union { unsigned short u; bf16 h; } b; b.u = s; return b.h;
}
static __device__ __forceinline__ v8bf ld8(const unsigned short* p) {
  return *(const v8bf*)p;
}
static __device__ __forceinline__ v8f wmma_bf16(v16bf a, v16bf b, v8f c) {
  // D = A(16x32) * B(32x16) + C, f32 accumulate
  return __builtin_amdgcn_wmma_f32_16x16x32_bf16(false, a, false, b, (short)0, c, false, false);
}

// ---- CDNA5 async memory->LDS copy (ASYNCcnt-tracked, no VGPR round trip) ----
// Generic pointers into the LDS aperture carry the LDS byte offset in their
// low 32 bits, so truncation gives the per-lane LDS destination address.
static __device__ __forceinline__ void async_ld_b128(void* lds_ptr, const void* gptr) {
  unsigned lds_addr = (unsigned)(size_t)lds_ptr;
  unsigned long long ga = (unsigned long long)(size_t)gptr;
  asm volatile("global_load_async_to_lds_b128 %0, %1, off"
               :: "v"(lds_addr), "v"(ga) : "memory");
}
static __device__ __forceinline__ void wait_async0() {
  asm volatile("s_wait_asynccnt 0x0" ::: "memory");
}

// ---------------- elementwise f32 -> bf16 weight convert ----------------
__global__ void cvt_k(const float* __restrict__ s, unsigned short* __restrict__ d, size_t n) {
  size_t i = (size_t)blockIdx.x * blockDim.x + threadIdx.x;
  if (i < n) d[i] = f2bf(s[i]);
}

// ---------------- per-sample stable compaction of visible tokens ----------------
__global__ void scan_k(const int* __restrict__ mask, int* __restrict__ order,
                       int* __restrict__ lens, int Sp) {
  int b = blockIdx.x;
  if (threadIdx.x == 0) {
    int cnt = 0;
    for (int t = 0; t < T_; ++t)
      if (mask[b * T_ + t] != 0) { if (cnt < Sp) order[b * Sp + cnt] = t; ++cnt; }
    if (cnt > Sp) cnt = Sp;
    lens[b] = cnt;
    for (int i = cnt; i < Sp; ++i) order[b * Sp + i] = 0;
  }
}

// ---------------- im2col of gathered patches -> bf16 A matrix [B*Sp][768] ----------------
__global__ void im2col_k(const float* __restrict__ x, const int* __restrict__ order,
                         const int* __restrict__ lens, unsigned short* __restrict__ Ap, int Sp) {
  size_t idx = (size_t)blockIdx.x * blockDim.x + threadIdx.x;
  size_t total = (size_t)B_ * Sp * 768;
  if (idx >= total) return;
  int kk = (int)(idx % 768);
  size_t mi = idx / 768;
  int b = (int)(mi / Sp), i = (int)(mi % Sp);
  float v = 0.f;
  if (i < lens[b]) {
    int t  = order[mi];
    int c  = kk >> 8, py = (kk >> 4) & 15, px = kk & 15;
    int gy = t / G_, gx = t % G_;
    v = x[(((size_t)b * C_ + c) * IMG_ + (size_t)gy * P_ + py) * IMG_ + gx * P_ + px];
  }
  Ap[idx] = f2bf(v);
}

// ---------------- generic bf16 WMMA GEMM: out[M][N] = A[M][K] * W[N][K]^T (+bias) ----------------
// EPI: 0 = f32 out, 1 = bf16 out, 2 = bf16 out + exact GELU, 3 = patch-embed epilogue
#define TM 128
#define TN 128
#define TK 32
#define LDT 40   // 32 + 8 pad, keeps 16B alignment of fragment reads

template<int EPI>
__global__ __launch_bounds__(256) void gemm_bf16_k(
    const unsigned short* __restrict__ A, const unsigned short* __restrict__ W,
    const float* __restrict__ bias, float* __restrict__ outF, unsigned short* __restrict__ outB,
    const int* __restrict__ order, const int* __restrict__ lens, const float* __restrict__ pos,
    int M, int N, int K, int Sp)
{
  __shared__ __align__(16) unsigned short As[TM * LDT];
  __shared__ __align__(16) unsigned short Bs[TN * LDT];
  const int tid  = threadIdx.x;
  const int lane = tid & 31, wid = tid >> 5;
  const int waveM = (wid & 3) * 32;     // 4 waves along M
  const int waveN = (wid >> 2) * 64;    // 2 waves along N
  const int mBlk = blockIdx.y * TM;
  const int nBlk = blockIdx.x * TN;
  const int r15 = lane & 15;
  const int koA = (lane >> 4) * 16;     // contiguous k = 16*(lane/16) + s, same order as B
  const int koB = (lane >> 4) * 16;     // contiguous k = 16*(lane/16) + s, same order as A

  v8f acc[2][4];
  #pragma unroll
  for (int fm = 0; fm < 2; ++fm)
    #pragma unroll
    for (int fn = 0; fn < 4; ++fn)
      #pragma unroll
      for (int v = 0; v < 8; ++v) acc[fm][fn][v] = 0.f;

  for (int k0 = 0; k0 < K; k0 += TK) {
    __syncthreads();
    // Async tile staging: 512 chunks of 8 bf16 per tile, direct memory->LDS
    #pragma unroll
    for (int c = 0; c < 2; ++c) {
      int idx = tid + c * 256;
      int row = idx >> 2, seg = (idx & 3) * 8;
      async_ld_b128(&As[row * LDT + seg], A + (size_t)(mBlk + row) * K + k0 + seg);
      async_ld_b128(&Bs[row * LDT + seg], W + (size_t)(nBlk + row) * K + k0 + seg);
    }
    wait_async0();
    __syncthreads();
    v16bf af[2];
    #pragma unroll
    for (int fm = 0; fm < 2; ++fm) {
      const unsigned short* p = &As[(waveM + fm * 16 + r15) * LDT + koA];
      af[fm] = cat8(ld8(p), ld8(p + 8));
    }
    v16bf bfr[4];
    #pragma unroll
    for (int fn = 0; fn < 4; ++fn) {
      const unsigned short* p = &Bs[(waveN + fn * 16 + r15) * LDT + koB];
      bfr[fn] = cat8(ld8(p), ld8(p + 8));
    }
    #pragma unroll
    for (int fm = 0; fm < 2; ++fm)
      #pragma unroll
      for (int fn = 0; fn < 4; ++fn)
        acc[fm][fn] = wmma_bf16(af[fm], bfr[fn], acc[fm][fn]);
  }

  #pragma unroll
  for (int fm = 0; fm < 2; ++fm) {
    #pragma unroll
    for (int fn = 0; fn < 4; ++fn) {
      #pragma unroll
      for (int v = 0; v < 8; ++v) {
        int m = mBlk + waveM + fm * 16 + v + (lane >> 4) * 8;  // ISA C/D lane layout
        int n = nBlk + waveN + fn * 16 + r15;
        float val = acc[fm][fn][v] + bias[n];
        size_t o = (size_t)m * N + n;
        if (EPI == 0) {
          outF[o] = val;
        } else if (EPI == 1) {
          outB[o] = f2bf(val);
        } else if (EPI == 2) {
          float g = 0.5f * val * (1.0f + erff(val * 0.70710678118f));
          outB[o] = f2bf(g);
        } else {                           // patch embed: +pos_embed, zero pads
          int b = m / Sp, i = m - b * Sp;
          float out = 0.f;
          if (i < lens[b]) out = val + pos[(size_t)order[m] * D_ + n];
          outF[o] = out;
          outB[o] = f2bf(out);
        }
      }
    }
  }
}

// ---------------- attention: one wave per (batch, head, 16-query tile) ----------------
__global__ __launch_bounds__(32) void attn_k(
    const unsigned short* __restrict__ qkv,   // [B*Sp][3D], Q|K|V each D, head-major inside
    unsigned short* __restrict__ aout,        // [B*Sp][D]  bf16
    const int* __restrict__ lens, int Sp)
{
  __shared__ __align__(16) float sc[16 * SMAX];
  __shared__ __align__(16) unsigned short pb[16 * SMAX];
  const int b = blockIdx.x, h = blockIdx.y, mt = blockIdx.z;
  const int lane = threadIdx.x;
  const int r15 = lane & 15;
  const int m0 = mt * 16;
  const int len = lens[b];
  const size_t rs = 3 * D_;
  const unsigned short* base = qkv + (size_t)b * Sp * rs;

  // Q fragments: 16x64 -> two 16x32 A-frags
  v16bf qf[2];
  {
    const unsigned short* qrow = base + (size_t)(m0 + r15) * rs + h * HD_ + (lane >> 4) * 16;
    #pragma unroll
    for (int kk = 0; kk < 2; ++kk) {
      const unsigned short* p = qrow + kk * 32;
      qf[kk] = cat8(ld8(p), ld8(p + 8));
    }
  }
  const int koB = (lane >> 4) * 16;

  // scores = (Q K^T) * 1/sqrt(64), key-pad mask
  for (int kt = 0; kt < Sp / 16; ++kt) {
    v8f s;
    #pragma unroll
    for (int v = 0; v < 8; ++v) s[v] = 0.f;
    const unsigned short* krow = base + (size_t)(kt * 16 + r15) * rs + D_ + h * HD_ + koB;
    #pragma unroll
    for (int kk = 0; kk < 2; ++kk) {
      const unsigned short* p = krow + kk * 32;
      v16bf kf = cat8(ld8(p), ld8(p + 8));     // B = K^T, column = key token
      s = wmma_bf16(qf[kk], kf, s);
    }
    int key = kt * 16 + r15;
    bool vis = key < len;
    #pragma unroll
    for (int v = 0; v < 8; ++v) {
      int mr = v + (lane >> 4) * 8;
      sc[mr * SMAX + key] = vis ? s[v] * 0.125f : -1.0e30f;
    }
  }
  __syncthreads();

  // rowwise softmax -> bf16 P in LDS
  if (lane < 16) {
    float mx = -3.0e38f;
    for (int j = 0; j < Sp; ++j) mx = fmaxf(mx, sc[lane * SMAX + j]);
    float sum = 0.f;
    for (int j = 0; j < Sp; ++j) {
      float e = __expf(sc[lane * SMAX + j] - mx);
      sc[lane * SMAX + j] = e;
      sum += e;
    }
    float inv = 1.0f / sum;
    for (int j = 0; j < Sp; ++j) pb[lane * SMAX + j] = f2bf(sc[lane * SMAX + j] * inv);
  }
  __syncthreads();

  // O = P V
  v8f o[4];
  #pragma unroll
  for (int dt = 0; dt < 4; ++dt)
    #pragma unroll
    for (int v = 0; v < 8; ++v) o[dt][v] = 0.f;
  for (int ct = 0; ct < Sp / 32; ++ct) {
    const unsigned short* pp = &pb[r15 * SMAX + ct * 32 + (lane >> 4) * 16];
    v16bf pf = cat8(ld8(pp), ld8(pp + 8));     // A-frag of P (16x32)
    #pragma unroll
    for (int dt = 0; dt < 4; ++dt) {
      int d = h * HD_ + dt * 16 + r15;
      const unsigned short* vcol = base + 2 * D_ + d;
      int kb = ct * 32 + (lane >> 4) * 16;
      v16bf vf;
      #pragma unroll
      for (int j = 0; j < 16; ++j)
        vf[j] = bfbits(vcol[(size_t)(kb + j) * rs]);  // B[k=key][n=d], strided gather
      o[dt] = wmma_bf16(pf, vf, o[dt]);
    }
  }
  #pragma unroll
  for (int dt = 0; dt < 4; ++dt)
    #pragma unroll
    for (int v = 0; v < 8; ++v) {
      int mr = m0 + v + (lane >> 4) * 8;
      int d  = h * HD_ + dt * 16 + r15;
      aout[(size_t)(b * Sp + mr) * D_ + d] = f2bf(o[dt][v]);
    }
}

// ---------------- residual add + LayerNorm (f32 master, bf16 mirror) ----------------
__global__ __launch_bounds__(256) void resln_k(
    float* __restrict__ tok, const float* __restrict__ p, unsigned short* __restrict__ xbf,
    const float* __restrict__ g, const float* __restrict__ bta)
{
  __shared__ float red[256];
  const size_t row = blockIdx.x;
  const int tid = threadIdx.x;
  float loc[3];
  float s = 0.f;
  #pragma unroll
  for (int it = 0; it < 3; ++it) {
    int j = tid + it * 256;
    float v = tok[row * D_ + j] + p[row * D_ + j];
    loc[it] = v; s += v;
  }
  red[tid] = s; __syncthreads();
  for (int o = 128; o > 0; o >>= 1) { if (tid < o) red[tid] += red[tid + o]; __syncthreads(); }
  float mean = red[0] * (1.0f / (float)D_);
  __syncthreads();
  s = 0.f;
  #pragma unroll
  for (int it = 0; it < 3; ++it) { float d = loc[it] - mean; s += d * d; }
  red[tid] = s; __syncthreads();
  for (int o = 128; o > 0; o >>= 1) { if (tid < o) red[tid] += red[tid + o]; __syncthreads(); }
  float rstd = rsqrtf(red[0] * (1.0f / (float)D_) + EPS_);
  #pragma unroll
  for (int it = 0; it < 3; ++it) {
    int j = tid + it * 256;
    float v = (loc[it] - mean) * rstd * g[j] + bta[j];
    tok[row * D_ + j] = v;
    xbf[row * D_ + j] = f2bf(v);
  }
}

// ---------------- final LayerNorm -> d_out ctx + pad flags ----------------
__global__ __launch_bounds__(256) void final_k(
    const float* __restrict__ tok, const float* __restrict__ g, const float* __restrict__ bta,
    const int* __restrict__ lens, float* __restrict__ out, int Sc, int Sp)
{
  __shared__ float red[256];
  const int b = blockIdx.x / Sc;
  const int i = blockIdx.x % Sc;
  const size_t row = (size_t)b * Sp + i;
  const int tid = threadIdx.x;
  float loc[3];
  float s = 0.f;
  #pragma unroll
  for (int it = 0; it < 3; ++it) {
    int j = tid + it * 256;
    float v = tok[row * D_ + j];
    loc[it] = v; s += v;
  }
  red[tid] = s; __syncthreads();
  for (int o = 128; o > 0; o >>= 1) { if (tid < o) red[tid] += red[tid + o]; __syncthreads(); }
  float mean = red[0] * (1.0f / (float)D_);
  __syncthreads();
  s = 0.f;
  #pragma unroll
  for (int it = 0; it < 3; ++it) { float d = loc[it] - mean; s += d * d; }
  red[tid] = s; __syncthreads();
  for (int o = 128; o > 0; o >>= 1) { if (tid < o) red[tid] += red[tid + o]; __syncthreads(); }
  float rstd = rsqrtf(red[0] * (1.0f / (float)D_) + EPS_);
  #pragma unroll
  for (int it = 0; it < 3; ++it) {
    int j = tid + it * 256;
    out[((size_t)b * Sc + i) * D_ + j] = (loc[it] - mean) * rstd * g[j] + bta[j];
  }
  if (tid == 0)
    out[(size_t)B_ * Sc * D_ + (size_t)b * Sc + i] = (i >= lens[b]) ? 1.0f : 0.0f;
}

// ---------------- host orchestration ----------------
extern "C" void kernel_launch(void* const* d_in, const int* in_sizes, int n_in,
                              void* d_out, int out_size, void* d_ws, size_t ws_size,
                              hipStream_t stream)
{
  const float* x       = (const float*)d_in[0];
  const int*   cmask   = (const int*)  d_in[1];
  const float* patch_w = (const float*)d_in[2];
  const float* patch_b = (const float*)d_in[3];
  const float* pos     = (const float*)d_in[4];
  const float* wqkv    = (const float*)d_in[5];
  const float* bqkv    = (const float*)d_in[6];
  const float* wo      = (const float*)d_in[7];
  const float* bo      = (const float*)d_in[8];
  const float* ln1g    = (const float*)d_in[9];
  const float* ln1b    = (const float*)d_in[10];
  const float* w1      = (const float*)d_in[11];
  const float* b1      = (const float*)d_in[12];
  const float* w2      = (const float*)d_in[13];
  const float* b2      = (const float*)d_in[14];
  const float* ln2g    = (const float*)d_in[15];
  const float* ln2b    = (const float*)d_in[16];
  const float* ng      = (const float*)d_in[17];
  const float* nb      = (const float*)d_in[18];
  (void)in_sizes; (void)n_in; (void)ws_size;

  // Sc from out_size = B*Sc*(D+1); pad to multiple of 32 for WMMA tiling
  int Sc = out_size / (B_ * (D_ + 1));
  if (Sc < 1) Sc = 1;
  if (Sc > T_) Sc = T_;
  int Sp = (Sc + 31) & ~31;
  if (Sp > SMAX) Sp = SMAX;
  const int M = B_ * Sp;

  // carve workspace
  char* wp = (char*)d_ws;
  auto alloc = [&](size_t bytes) -> char* {
    char* r = wp; wp += (bytes + 255) & ~(size_t)255; return r;
  };
  unsigned short* bf_qkvw = (unsigned short*)alloc(2ull * NL_ * 3 * D_ * D_);
  unsigned short* bf_wo   = (unsigned short*)alloc(2ull * NL_ * D_ * D_);
  unsigned short* bf_w1   = (unsigned short*)alloc(2ull * NL_ * 4 * D_ * D_);
  unsigned short* bf_w2   = (unsigned short*)alloc(2ull * NL_ * 4 * D_ * D_);
  unsigned short* bf_pw   = (unsigned short*)alloc(2ull * D_ * (C_ * P_ * P_));
  int*            order   = (int*)           alloc(4ull * M);
  int*            lens    = (int*)           alloc(4ull * B_);
  unsigned short* Ap      = (unsigned short*)alloc(2ull * M * 768);
  float*          tok     = (float*)         alloc(4ull * M * D_);
  unsigned short* xbf     = (unsigned short*)alloc(2ull * M * D_);
  unsigned short* qkvbf   = (unsigned short*)alloc(2ull * M * 3 * D_);
  unsigned short* abf     = (unsigned short*)alloc(2ull * M * D_);
  float*          pf      = (float*)         alloc(4ull * M * D_);
  unsigned short* hbf     = (unsigned short*)alloc(2ull * M * 4 * D_);

  auto cvt = [&](const float* s, unsigned short* d, size_t n) {
    cvt_k<<<(unsigned)((n + 255) / 256), 256, 0, stream>>>(s, d, n);
  };
  cvt(wqkv,    bf_qkvw, (size_t)NL_ * 3 * D_ * D_);
  cvt(wo,      bf_wo,   (size_t)NL_ * D_ * D_);
  cvt(w1,      bf_w1,   (size_t)NL_ * 4 * D_ * D_);
  cvt(w2,      bf_w2,   (size_t)NL_ * 4 * D_ * D_);
  cvt(patch_w, bf_pw,   (size_t)D_ * (C_ * P_ * P_));

  scan_k<<<B_, 32, 0, stream>>>(cmask, order, lens, Sp);

  {
    size_t total = (size_t)M * 768;
    im2col_k<<<(unsigned)((total + 255) / 256), 256, 0, stream>>>(x, order, lens, Ap, Sp);
  }

  // patch embedding GEMM: [M,768] x [768,768]^T + bias + pos, zero pads
  gemm_bf16_k<3><<<dim3(D_ / TN, M / TM), 256, 0, stream>>>(
      Ap, bf_pw, patch_b, tok, xbf, order, lens, pos, M, D_, 768, Sp);

  for (int l = 0; l < NL_; ++l) {
    // QKV projection -> bf16
    gemm_bf16_k<1><<<dim3((3 * D_) / TN, M / TM), 256, 0, stream>>>(
        xbf, bf_qkvw + (size_t)l * 3 * D_ * D_, bqkv + (size_t)l * 3 * D_,
        nullptr, qkvbf, nullptr, nullptr, nullptr, M, 3 * D_, D_, Sp);
    // attention
    attn_k<<<dim3(B_, NH_, Sp / 16), 32, 0, stream>>>(qkvbf, abf, lens, Sp);
    // out projection -> f32
    gemm_bf16_k<0><<<dim3(D_ / TN, M / TM), 256, 0, stream>>>(
        abf, bf_wo + (size_t)l * D_ * D_, bo + (size_t)l * D_,
        pf, nullptr, nullptr, nullptr, nullptr, M, D_, D_, Sp);
    // x = LN(x + attn)
    resln_k<<<M, 256, 0, stream>>>(tok, pf, xbf, ln1g + (size_t)l * D_, ln1b + (size_t)l * D_);
    // FFN up + GELU -> bf16
    gemm_bf16_k<2><<<dim3((4 * D_) / TN, M / TM), 256, 0, stream>>>(
        xbf, bf_w1 + (size_t)l * 4 * D_ * D_, b1 + (size_t)l * 4 * D_,
        nullptr, hbf, nullptr, nullptr, nullptr, M, 4 * D_, D_, Sp);
    // FFN down -> f32
    gemm_bf16_k<0><<<dim3(D_ / TN, M / TM), 256, 0, stream>>>(
        hbf, bf_w2 + (size_t)l * D_ * 4 * D_, b2 + (size_t)l * D_,
        pf, nullptr, nullptr, nullptr, nullptr, M, D_, 4 * D_, Sp);
    // x = LN(x + ffn)
    resln_k<<<M, 256, 0, stream>>>(tok, pf, xbf, ln2g + (size_t)l * D_, ln2b + (size_t)l * D_);
  }

  final_k<<<B_ * Sc, 256, 0, stream>>>(tok, ng, nb, lens, (float*)d_out, Sc, Sp);
}
